// BaselineDNN_44409961841014
// MI455X (gfx1250) — matrix operand order repaired and checked by reading, not compiled, not run on hardware
//
#include <hip/hip_runtime.h>
#include <hip/hip_bf16.h>
#include <hip/hip_fp16.h>

typedef __attribute__((ext_vector_type(16))) _Float16 v16h;
typedef __attribute__((ext_vector_type(8)))  _Float16 v8h;
typedef __attribute__((ext_vector_type(4)))  _Float16 v4h;
typedef __attribute__((ext_vector_type(8)))  float    v8f;

#define DIM   256   // embedding dim
#define HDIM  128   // hidden dim (DIM/2)
#define BATCH 8192
#define SEQ   128

// ---------------------------------------------------------------------------
// Kernel 1: convert W1 [HDIM, DIM] f32 -> f16 (one-shot, 32K elements)
// ---------------------------------------------------------------------------
__global__ void cvt_w1_kernel(const float* __restrict__ W1,
                              _Float16* __restrict__ W1h, int n) {
    int i = blockIdx.x * blockDim.x + threadIdx.x;
    if (i < n) W1h[i] = (_Float16)W1[i];
}

// ---------------------------------------------------------------------------
// Kernel 2: ragged-mean embedding pooling.
// One block per batch row. 4 token-groups x 64 lanes of float4 loads
// (emb_table is L2-resident: 102.4MB < 192MB). Result written as f16.
// ---------------------------------------------------------------------------
__global__ __launch_bounds__(256) void pool_kernel(
    const int* __restrict__ x, const int* __restrict__ lengths,
    const float* __restrict__ emb, _Float16* __restrict__ repH) {
    const int b   = blockIdx.x;
    const int tid = threadIdx.x;
    const int g   = tid >> 6;   // token group 0..3
    const int t   = tid & 63;   // float4 slot: covers 256 floats
    const int len = lengths[b];

    const int* xb = x + b * SEQ;
    float4 acc = make_float4(0.f, 0.f, 0.f, 0.f);
    for (int l = g; l < len; l += 4) {
        const int idx = xb[l];
        const float4 v = ((const float4*)(emb + (size_t)idx * DIM))[t];
        acc.x += v.x; acc.y += v.y; acc.z += v.z; acc.w += v.w;
    }

    __shared__ float4 red[256];
    red[tid] = acc;
    __syncthreads();
    if (g == 0) {
        const float4 a0 = red[t];
        const float4 a1 = red[64 + t];
        const float4 a2 = red[128 + t];
        const float4 a3 = red[192 + t];
        const float s = 1.0f / (float)len;
        v4h o;
        o[0] = (_Float16)((a0.x + a1.x + a2.x + a3.x) * s);
        o[1] = (_Float16)((a0.y + a1.y + a2.y + a3.y) * s);
        o[2] = (_Float16)((a0.z + a1.z + a2.z + a3.z) * s);
        o[3] = (_Float16)((a0.w + a1.w + a2.w + a3.w) * s);
        *(v4h*)(repH + (size_t)b * DIM + 4 * t) = o;
    }
}

// ---------------------------------------------------------------------------
// Kernel 3: fused  h = relu(rep @ W1^T + b1); out = sigmoid(h @ W2^T + b2)
// 512 blocks x 8 waves. Wave w computes the 16x16 tile (mTile, nTile=w) of h
// via 8x v_wmma_f32_16x16x32_f16 along K=256, then fuses the W2 dot product.
// ---------------------------------------------------------------------------
__global__ __launch_bounds__(256) void mlp_kernel(
    const _Float16* __restrict__ repH, const _Float16* __restrict__ W1h,
    const float* __restrict__ b1, const float* __restrict__ W2,
    const float* __restrict__ b2, float* __restrict__ out) {
    const int tid   = threadIdx.x;
    const int wave  = tid >> 5;       // nTile 0..7  (covers N = 128)
    const int lane  = tid & 31;
    const int grp   = lane >> 4;      // half-wave (K-group select)
    const int ln    = lane & 15;
    const int mTile = blockIdx.x;

    const int m = mTile * 16 + ln;    // A-matrix row held by this lane
    const int n = wave * 16 + ln;     // B-matrix col / output column

    const _Float16* arow = repH + (size_t)m * DIM;
    const _Float16* brow = W1h  + (size_t)n * DIM;

    v8f c = {};
    #pragma unroll
    for (int kk = 0; kk < DIM; kk += 32) {
        // A 16x32 f16 layout: lane holds K = kk + 8*grp + {0..7} and +16
        union { v16h v; v8h h[2]; } A;
        A.h[0] = *(const v8h*)(arow + kk + grp * 8);
        A.h[1] = *(const v8h*)(arow + kk + grp * 8 + 16);
        // B 32x16 f16 layout: lane holds K = kk + 8*grp + {0..7} and +16
        union { v16h v; v8h h[2]; } Bm;
        Bm.h[0] = *(const v8h*)(brow + kk + grp * 8);
        Bm.h[1] = *(const v8h*)(brow + kk + grp * 8 + 16);
        c = __builtin_amdgcn_wmma_f32_16x16x32_f16(
                false, A.v, false, Bm.v, (short)0, c, false, false);
    }

    // Epilogue: bias + relu, scale by W2[n], reduce 16 lanes per row.
    const float bias = b1[n];
    const float w2v  = W2[n];
    float p[8];
    #pragma unroll
    for (int r = 0; r < 8; ++r) {
        float h = c[r] + bias;            // row = grp*8 + r, col = n
        h = h > 0.f ? h : 0.f;
        p[r] = h * w2v;
    }
    #pragma unroll
    for (int r = 0; r < 8; ++r) {
        float v = p[r];
        v += __shfl_xor(v, 1, 32);
        v += __shfl_xor(v, 2, 32);
        v += __shfl_xor(v, 4, 32);
        v += __shfl_xor(v, 8, 32);        // masks <16: stays within half-wave
        p[r] = v;
    }

    __shared__ float sAcc[16];
    if (tid < 16) sAcc[tid] = 0.f;
    __syncthreads();
    if (ln == 0) {
        #pragma unroll
        for (int r = 0; r < 8; ++r)
            atomicAdd(&sAcc[grp * 8 + r], p[r]);   // ds_add_f32
    }
    __syncthreads();
    if (tid < 16) {
        const float z = sAcc[tid] + b2[0];
        out[mTile * 16 + tid] = 1.f / (1.f + __expf(-z));
    }
}

// ---------------------------------------------------------------------------
extern "C" void kernel_launch(void* const* d_in, const int* in_sizes, int n_in,
                              void* d_out, int out_size, void* d_ws, size_t ws_size,
                              hipStream_t stream) {
    const int*   x       = (const int*)d_in[0];
    const int*   lengths = (const int*)d_in[1];
    const float* emb     = (const float*)d_in[2];
    const float* W1      = (const float*)d_in[3];
    const float* b1      = (const float*)d_in[4];
    const float* W2      = (const float*)d_in[5];
    const float* b2      = (const float*)d_in[6];
    float*       out     = (float*)d_out;

    _Float16* repH = (_Float16*)d_ws;                                   // 4 MB
    _Float16* W1h  = (_Float16*)((char*)d_ws + (size_t)BATCH * DIM * 2); // 64 KB

    cvt_w1_kernel<<<(HDIM * DIM + 255) / 256, 256, 0, stream>>>(W1, W1h, HDIM * DIM);
    pool_kernel  <<<BATCH, 256, 0, stream>>>(x, lengths, emb, repH);
    mlp_kernel   <<<BATCH / 16, 256, 0, stream>>>(repH, W1h, b1, W2, b2, out);
}
